// lastgcn_76390288327744
// MI455X (gfx1250) — matrix operand order repaired and checked by reading, not compiled, not run on hardware
//
#include <hip/hip_runtime.h>
#include <hip/hip_bf16.h>

typedef __attribute__((ext_vector_type(16))) _Float16 v16h;
typedef __attribute__((ext_vector_type(8)))  float    v8f;

#define NNODES 588
#define NPAD   592          // 37 tiles of 16 rows
#define NEDGE  9408
#define HID    64
#define NTHREADS 512        // 16 wave32
#define NEGS   0.01f
#define THRESHV 0.999f

__device__ __forceinline__ float lrelu(float v)    { return v >= 0.f ? v : NEGS * v; }
__device__ __forceinline__ float sigmoidf(float v) { return 1.f / (1.f + __expf(-v)); }

__device__ __forceinline__ void cvt8(v16h& a, int base, float4 lo, float4 hi) {
  a[base + 0] = (_Float16)lo.x; a[base + 1] = (_Float16)lo.y;
  a[base + 2] = (_Float16)lo.z; a[base + 3] = (_Float16)lo.w;
  a[base + 4] = (_Float16)hi.x; a[base + 5] = (_Float16)hi.y;
  a[base + 6] = (_Float16)hi.z; a[base + 7] = (_Float16)hi.w;
}

// agg[d][c] = sum over edges e with dst[e]==d of xs[src[e]][c]   (all in LDS)
// Thread owns one fixed channel c, walks edges with stride 8: consecutive lanes
// hit consecutive LDS addresses -> conflict-free ds_add_f32.
__device__ __forceinline__ void aggregate64(int tid,
                                            const float* __restrict__ xs,
                                            float* __restrict__ agg,
                                            const int* __restrict__ src,
                                            const int* __restrict__ dst) {
  for (int i = tid; i < NPAD * HID; i += NTHREADS) agg[i] = 0.f;
  __syncthreads();
  const int c = tid & 63;
  for (int e = (tid >> 6); e < NEDGE; e += (NTHREADS >> 6)) {
    atomicAdd(&agg[dst[e] * HID + c], xs[src[e] * HID + c]);
  }
  __syncthreads();
}

// xs = lrelu(agg @ W2 + b2), 588x64 x 64x64, via v_wmma_f32_16x16x32_f16.
// wave/mStart are SCALAR (readfirstlane) -> uniform loop, EXEC stays all-ones.
// B operands (wb0,wb1) and bias are hoisted into registers by the caller.
__device__ __forceinline__ void gemm_w2_wmma(int mStart, int lane,
                                             const float* __restrict__ agg,
                                             float* __restrict__ xs,
                                             v16h wb0, v16h wb1,
                                             float bias, int col) {
  const int hi16 = lane >> 4;          // 0: lanes 0-15, 1: lanes 16-31
  const int l15  = lane & 15;
  for (int m = mStart; m < NPAD / 16; m += 4) {
    const int row = m * 16 + l15;
    const float* base = agg + row * HID + hi16 * 8;
    // A 16x32 f16 layout: lane<16: K{0..7,16..23}; lane>=16: K{8..15,24..31}
    // -> two contiguous float4 pairs per operand (ds_load_b128)
    v16h a0, a1;
    cvt8(a0, 0, *(const float4*)(base +  0), *(const float4*)(base +  4));
    cvt8(a0, 8, *(const float4*)(base + 16), *(const float4*)(base + 20));
    cvt8(a1, 0, *(const float4*)(base + 32), *(const float4*)(base + 36));
    cvt8(a1, 8, *(const float4*)(base + 48), *(const float4*)(base + 52));
    v8f c = {0.f, 0.f, 0.f, 0.f, 0.f, 0.f, 0.f, 0.f};
    c = __builtin_amdgcn_wmma_f32_16x16x32_f16(false, a0, false, wb0, (short)0, c, false, false);
    c = __builtin_amdgcn_wmma_f32_16x16x32_f16(false, a1, false, wb1, (short)0, c, false, false);
#pragma unroll
    for (int v = 0; v < 8; ++v) {
      const int orow = m * 16 + hi16 * 8 + v;        // C/D: VGPR v -> row v (+8 hi lanes)
      xs[orow * HID + col] = lrelu(c[v] + bias);
    }
  }
}

__global__ __launch_bounds__(NTHREADS, 1)
void lastgcn_fused(const float* __restrict__ features,
                   const int* __restrict__ src, const int* __restrict__ dst,
                   const float* __restrict__ W1, const float* __restrict__ b1,
                   const float* __restrict__ W2, const float* __restrict__ b2,
                   const float* __restrict__ W3, const float* __restrict__ b3,
                   const float* __restrict__ W5, const float* __restrict__ b5,
                   const float* __restrict__ gwi, const float* __restrict__ gwh,
                   const float* __restrict__ gbi, const float* __restrict__ gbh,
                   const float* __restrict__ w_end, const float* __restrict__ b_end,
                   const int* __restrict__ capp,
                   float* __restrict__ out) {
  extern __shared__ float sm[];
  float* xs     = sm;                        // 592*64
  float* agg    = xs + NPAD * HID;           // 592*64  (16B aligned)
  float* inputt = agg + NPAD * HID;          // 592
  float* gx     = inputt + NPAD;             // 192
  float* gh     = gx + 192;                  // 192
  float* state  = gh + 192;                  // 64 (frozen after pre-loop update)
  float* stn    = state + 64;                // 64
  float* scal   = stn + 64;                  // [0]=s [1]=fprob [2]=counter
  _Float16* w2h = (_Float16*)(scal + 16);    // 64*64 f16

  const int tid  = threadIdx.x;
  const int lane = tid & 31;
  const int wave = __builtin_amdgcn_readfirstlane(tid >> 5);   // scalar wave id
  const int cap  = capp[0];

  // stage W2 -> LDS f16 (row-major)
  for (int i = tid; i < HID * HID; i += NTHREADS) w2h[i] = (_Float16)W2[i];

  // ---- conv1: agg4 = segment_sum(features); x = lrelu(agg4 @ W1 + b1)
  for (int i = tid; i < NPAD * 4; i += NTHREADS) agg[i] = 0.f;
  __syncthreads();

  // Hoist this wave's B operands (W2 N-tile) + bias for the whole kernel.
  // B 32x16 f16: lane<16 holds K{0..7,16..23}, lane>=16 holds K{8..15,24..31} (per column).
  const int ncol   = (wave & 3) * 16 + (lane & 15);
  const int mStart = wave >> 2;                 // scalar
  v16h wb0, wb1;
  {
    const int hi16 = lane >> 4;
#pragma unroll
    for (int j = 0; j < 16; ++j) {
      wb0[j] = w2h[( 0 + hi16 * 8 + (j & 7) + (j >> 3) * 16) * HID + ncol];
      wb1[j] = w2h[(32 + hi16 * 8 + (j & 7) + (j >> 3) * 16) * HID + ncol];
    }
  }
  const float bias2 = b2[ncol];

  {
    const int c = tid & 3;
    for (int e = (tid >> 2); e < NEDGE; e += (NTHREADS >> 2))
      atomicAdd(&agg[dst[e] * 4 + c], features[src[e] * 4 + c]);
  }
  __syncthreads();
  for (int i = tid; i < NNODES * HID; i += NTHREADS) {
    const int d = i >> 6, j = i & 63;
    float acc = b1[j];
#pragma unroll
    for (int k = 0; k < 4; ++k) acc += agg[d * 4 + k] * W1[k * HID + j];
    xs[d * HID + j] = lrelu(acc);
  }
  __syncthreads();

  // ---- inputt0 = lrelu(conv(x, W3, b3)).T
  aggregate64(tid, xs, agg, src, dst);
  for (int d = tid; d < NNODES; d += NTHREADS) {
    float acc = b3[0];
    for (int c2 = 0; c2 < HID; ++c2) acc += agg[d * HID + c2] * W3[c2];
    inputt[d] = lrelu(acc);
  }
  if (tid < HID) state[tid] = 0.f;
  __syncthreads();

  // ---- gx = inputt @ gru_wi.T : one gate-row per wave, lanes coalesced along d,
  //      wave32 shuffle reduction. 12 rows per wave.
  for (int g = wave; g < 192; g += 16) {
    const float* wrow = gwi + g * NNODES;
    float acc = 0.f;
    for (int d = lane; d < NNODES; d += 32) acc += inputt[d] * wrow[d];
#pragma unroll
    for (int off = 16; off; off >>= 1) acc += __shfl_down(acc, off, 32);
    if (lane == 0) gx[g] = acc + gbi[g];
  }
  if (tid < 192) gh[tid] = gbh[tid];   // gh with h=0 is just the bias
  __syncthreads();
  if (tid < HID) {
    const float r  = sigmoidf(gx[tid] + gh[tid]);
    const float z  = sigmoidf(gx[HID + tid] + gh[HID + tid]);
    const float nn = tanhf(gx[2 * HID + tid] + r * gh[2 * HID + tid]);
    state[tid] = (1.f - z) * nn + z * state[tid];   // h==0 here
  }
  __syncthreads();
  // gh from the now-frozen state: constant across all loop iterations
  if (tid < 192) {
    const float* wrow = gwh + tid * HID;
    float acc = gbh[tid];
    for (int c2 = 0; c2 < HID; ++c2) acc += state[c2] * wrow[c2];
    gh[tid] = acc;
  }
  if (tid == 0) {
    float acc = b_end[0];
    for (int c2 = 0; c2 < HID; ++c2) acc += state[c2] * w_end[c2];
    const float fp = sigmoidf(acc);
    scal[0] = fp; scal[1] = fp; scal[2] = 0.f;
  }
  __syncthreads();

  // ---- scan: once s >= THRESH the carried state never changes again -> break
  for (int t = 0; t < cap; ++t) {
    __syncthreads();
    if (scal[0] >= THRESHV) break;

    aggregate64(tid, xs, agg, src, dst);                        // agg = segsum(x)
    gemm_w2_wmma(mStart, lane, agg, xs, wb0, wb1, bias2, ncol); // x = lrelu(agg@W2+b2)
    __syncthreads();

    aggregate64(tid, xs, agg, src, dst);                        // agg = segsum(x_new)
    for (int d = tid; d < NNODES; d += NTHREADS) {
      float acc = b3[0];
      for (int c2 = 0; c2 < HID; ++c2) acc += agg[d * HID + c2] * W3[c2];
      inputt[d] = lrelu(acc);
    }
    __syncthreads();

    for (int g = wave; g < 192; g += 16) {                      // coalesced GEMV
      const float* wrow = gwi + g * NNODES;
      float acc = 0.f;
      for (int d = lane; d < NNODES; d += 32) acc += inputt[d] * wrow[d];
#pragma unroll
      for (int off = 16; off; off >>= 1) acc += __shfl_down(acc, off, 32);
      if (lane == 0) gx[g] = acc + gbi[g];
    }
    __syncthreads();
    if (tid < HID) {                                 // gru with frozen state/gh
      const float r  = sigmoidf(gx[tid] + gh[tid]);
      const float z  = sigmoidf(gx[HID + tid] + gh[HID + tid]);
      const float nn = tanhf(gx[2 * HID + tid] + r * gh[2 * HID + tid]);
      stn[tid] = (1.f - z) * nn + z * state[tid];
    }
    __syncthreads();
    if (tid == 0) {
      float acc = b_end[0];
      for (int c2 = 0; c2 < HID; ++c2) acc += stn[c2] * w_end[c2];
      const float pr = sigmoidf(acc);
      scal[1] = pr;       // fprob
      scal[0] += pr;      // s
      scal[2] += 1.f;     // counter
    }
  }
  __syncthreads();

  // ---- out = conv(x_last, W5, b5)
  aggregate64(tid, xs, agg, src, dst);
  for (int i = tid; i < NNODES * 3; i += NTHREADS) {
    const int d = i / 3, j = i % 3;
    float acc = b5[j];
    for (int c2 = 0; c2 < HID; ++c2) acc += agg[d * HID + c2] * W5[c2 * 3 + j];
    out[i] = acc;
  }
  if (tid == 0) {
    out[NNODES * 3]     = scal[2];   // counter
    out[NNODES * 3 + 1] = scal[1];   // fprob
  }
}

extern "C" void kernel_launch(void* const* d_in, const int* in_sizes, int n_in,
                              void* d_out, int out_size, void* d_ws, size_t ws_size,
                              hipStream_t stream) {
  (void)in_sizes; (void)n_in; (void)d_ws; (void)ws_size; (void)out_size;
  const float* features = (const float*)d_in[0];
  const int*   src      = (const int*)d_in[1];
  const int*   dst      = (const int*)d_in[2];
  const float* W1 = (const float*)d_in[3];  const float* b1 = (const float*)d_in[4];
  const float* W2 = (const float*)d_in[5];  const float* b2 = (const float*)d_in[6];
  const float* W3 = (const float*)d_in[7];  const float* b3 = (const float*)d_in[8];
  const float* W5 = (const float*)d_in[9];  const float* b5 = (const float*)d_in[10];
  const float* gwi = (const float*)d_in[11]; const float* gwh = (const float*)d_in[12];
  const float* gbi = (const float*)d_in[13]; const float* gbh = (const float*)d_in[14];
  const float* w_end = (const float*)d_in[15]; const float* b_end = (const float*)d_in[16];
  const int*   capp  = (const int*)d_in[17];
  float* out = (float*)d_out;

  const size_t smem_bytes =
      (size_t)(2 * NPAD * HID + NPAD + 192 + 192 + 64 + 64 + 16) * sizeof(float)
      + (size_t)(HID * HID) * sizeof(_Float16);   // ~308 KB of the WGP's 320 KB LDS

  lastgcn_fused<<<dim3(1), dim3(NTHREADS), smem_bytes, stream>>>(
      features, src, dst, W1, b1, W2, b2, W3, b3, W5, b5,
      gwi, gwh, gbi, gbh, w_end, b_end, capp, out);
}
